// ACMGCN_80298708566455
// MI455X (gfx1250) — matrix operand order repaired and checked by reading, not compiled, not run on hardware
//
#include <hip/hip_runtime.h>
#include <hip/hip_bf16.h>

// ---------------------------------------------------------------------------
// Types for CDNA5 WMMA (wave32, 16x16x32 bf16 -> f32)
// ---------------------------------------------------------------------------
typedef __attribute__((ext_vector_type(16))) __bf16       v16bf;
typedef __attribute__((ext_vector_type(8)))  float        v8f;
typedef __attribute__((ext_vector_type(4)))  unsigned int u32x4;

union Frag16 { v16bf v; u32x4 q[2]; };
union BStage { u32x4 q[2]; unsigned short s[16]; };

__device__ __forceinline__ unsigned short f2bf(float f) {
    unsigned int u = __float_as_uint(f);
    unsigned int r = u + 0x7FFFu + ((u >> 16) & 1u);   // round-to-nearest-even
    return (unsigned short)(r >> 16);
}

__device__ __forceinline__ float waveSum(float x) {
#pragma unroll
    for (int o = 16; o > 0; o >>= 1) x += __shfl_xor(x, o, 32);
    return x;
}
__device__ __forceinline__ float waveMax(float x) {
#pragma unroll
    for (int o = 16; o > 0; o >>= 1) x = fmaxf(x, __shfl_xor(x, o, 32));
    return x;
}

// ---------------------------------------------------------------------------
// Elementwise helpers
// ---------------------------------------------------------------------------
__global__ __launch_bounds__(256) void cvt_f32_bf16(const float* __restrict__ src,
                                                    unsigned short* __restrict__ dst,
                                                    long n) {
    long i = (long)blockIdx.x * 256 + threadIdx.x;
    if (i < n) dst[i] = f2bf(src[i]);
}

__global__ __launch_bounds__(256) void zero_f32(float* __restrict__ p, long n) {
    long i = (long)blockIdx.x * 256 + threadIdx.x;
    if (i < n) p[i] = 0.0f;
}

// Pack three K x Ncol f32 matrices into one K x (3*Ncol) bf16 matrix.
__global__ __launch_bounds__(256) void pack3_bf16(const float* __restrict__ A,
                                                  const float* __restrict__ B,
                                                  const float* __restrict__ C,
                                                  unsigned short* __restrict__ W,
                                                  int K, int Ncol) {
    long i = (long)blockIdx.x * 256 + threadIdx.x;
    long total = (long)K * 3 * Ncol;
    if (i >= total) return;
    int r = (int)(i / (3 * Ncol));
    int c = (int)(i % (3 * Ncol));
    float v;
    if      (c <     Ncol) v = A[(size_t)r * Ncol + c];
    else if (c < 2 * Ncol) v = B[(size_t)r * Ncol + (c -     Ncol)];
    else                   v = C[(size_t)r * Ncol + (c - 2 * Ncol)];
    W[i] = f2bf(v);
}

// ---------------------------------------------------------------------------
// bf16 WMMA GEMM: C[M x Nn] = A[M x K] * B[K x Nn]  (fp32 accumulate)
// Block: 256 threads (8 wave32), 128x128 tile.
// Double-buffered LDS + register prefetch: next K-slab's global loads are
// issued before the 8 WMMAs of the current slab; B fragments are pipelined
// two-deep so WMMAs never drain DScnt to zero.
// ---------------------------------------------------------------------------
__global__ __launch_bounds__(256, 2) void gemm_bf16_wmma(
    const unsigned short* __restrict__ A, const unsigned short* __restrict__ B,
    float* __restrict__ C, int M, int Nn, int K)
{
    __shared__ alignas(16) unsigned short sA[2][128][40];   // 2 x 128x32 (+8 pad)
    __shared__ alignas(16) unsigned short sB[2][32][136];   // 2 x 32x128 (+8 pad)

    const int tid  = threadIdx.x;
    const int wave = tid >> 5;
    const int lane = tid & 31;
    const int m0 = blockIdx.x * 128;
    const int n0 = blockIdx.y * 128;

    v8f acc[8];
#pragma unroll
    for (int i = 0; i < 8; ++i)
#pragma unroll
        for (int j = 0; j < 8; ++j) acc[i][j] = 0.0f;

    const int rA = tid >> 1;            // 0..127 : row in A tile
    const int cA = (tid & 1) * 16;      // 0/16   : col base in A tile
    const int rB = tid >> 3;            // 0..31  : row in B tile
    const int cB = (tid & 7) * 16;      // 0..112 : col base in B tile
    const int  mA     = m0 + rA;
    const bool aValid = (mA < M);
    const int  colB0  = n0 + cB;

    const int mFrag = 16 * wave + (lane & 15);
    const int kbFrag = (lane >> 4) * 8;

    auto loadTile = [&](int k0, u32x4& a0, u32x4& a1, BStage& b) {
        a0 = (u32x4){0u, 0u, 0u, 0u};
        a1 = (u32x4){0u, 0u, 0u, 0u};
        if (aValid) {
            const u32x4* g = reinterpret_cast<const u32x4*>(A + (size_t)mA * K + k0 + cA);
            a0 = g[0]; a1 = g[1];
        }
        const size_t rowOff = (size_t)(k0 + rB) * Nn;
        if (colB0 + 16 <= Nn) {
            const u32x4* g = reinterpret_cast<const u32x4*>(B + rowOff + colB0);
            b.q[0] = g[0]; b.q[1] = g[1];
        } else {
#pragma unroll
            for (int j = 0; j < 16; ++j)
                b.s[j] = (colB0 + j < Nn) ? B[rowOff + colB0 + j] : (unsigned short)0;
        }
    };
    auto storeTile = [&](int buf, const u32x4& a0, const u32x4& a1, const BStage& b) {
        *reinterpret_cast<u32x4*>(&sA[buf][rA][cA])     = a0;
        *reinterpret_cast<u32x4*>(&sA[buf][rA][cA + 8]) = a1;
        *reinterpret_cast<u32x4*>(&sB[buf][rB][cB])     = b.q[0];
        *reinterpret_cast<u32x4*>(&sB[buf][rB][cB + 8]) = b.q[1];
    };

    const int nk = K >> 5;
    u32x4 pa0, pa1;
    BStage pb;
    loadTile(0, pa0, pa1, pb);
    storeTile(0, pa0, pa1, pb);

    for (int i = 0; i < nk; ++i) {
        __syncthreads();                      // buf[cur] visible; prior reads done
        const int  cur  = i & 1;
        const bool more = (i + 1) < nk;
        if (more) loadTile((i + 1) << 5, pa0, pa1, pb);   // overlap with compute

        // A fragment: lane<16 -> K {0..7,16..23}; lane>=16 -> K {8..15,24..31}
        Frag16 fa;
        fa.q[0] = *reinterpret_cast<const u32x4*>(&sA[cur][mFrag][kbFrag]);
        fa.q[1] = *reinterpret_cast<const u32x4*>(&sA[cur][mFrag][kbFrag + 16]);

        // B fragments pipelined two-deep (N col = lane&15; K same as A fragment)
        Frag16 fb[2];
#pragma unroll
        for (int s = 0; s < 16; ++s)
            fb[0].v[s] = __builtin_bit_cast(__bf16,
                sB[cur][kbFrag + (s & 7) + ((s >> 3) << 4)][lane & 15]);
#pragma unroll
        for (int nt = 0; nt < 8; ++nt) {
            if (nt < 7) {
#pragma unroll
                for (int s = 0; s < 16; ++s)
                    fb[(nt + 1) & 1].v[s] = __builtin_bit_cast(__bf16,
                        sB[cur][kbFrag + (s & 7) + ((s >> 3) << 4)][(nt + 1) * 16 + (lane & 15)]);
            }
            acc[nt] = __builtin_amdgcn_wmma_f32_16x16x32_bf16(
                false, fa.v, false, fb[nt & 1].v, (short)0, acc[nt], false, false);
        }

        if (more) storeTile(1 - cur, pa0, pa1, pb);       // after compute issue
    }

    // store: VGPR i -> M = 8*(lane>>4)+i, N = lane&15
    const int mBase = m0 + 16 * wave + (lane >> 4) * 8;
    const int nCol  = n0 + (lane & 15);
#pragma unroll
    for (int nt = 0; nt < 8; ++nt) {
        const int col = nCol + nt * 16;
        if (col < Nn) {
#pragma unroll
            for (int i = 0; i < 8; ++i) {
                const int m = mBase + i;
                if (m < M) C[(size_t)m * Nn + col] = acc[nt][i];
            }
        }
    }
}

// ---------------------------------------------------------------------------
// Dual SpMM (segment-sum over edges): accL += val*xl[col], accH += val*xh[col]
// One wave per edge. FEATS = 128 (layer 1) or 64 (layer 2).
// ---------------------------------------------------------------------------
template <int FEATS>
__global__ __launch_bounds__(256) void spmm_dual(
    const int* __restrict__ eRow, const int* __restrict__ eCol,
    const float* __restrict__ eVal,
    const float* __restrict__ XW, int ldx, int offH,
    float* __restrict__ accL, float* __restrict__ accH, int E)
{
    const long t = (long)blockIdx.x * 256 + threadIdx.x;
    const int  e = (int)(t >> 5);
    if (e >= E) return;
    const int lane = (int)(t & 31);
    constexpr int FPL = FEATS / 32;

    const int   r = eRow[e];
    const int   c = eCol[e];
    const float v = eVal[e];

    const float* xl = XW + (size_t)c * ldx + lane * FPL;
    const float* xh = xl + offH;
    float* dl = accL + (size_t)r * FEATS + lane * FPL;
    float* dh = accH + (size_t)r * FEATS + lane * FPL;
#pragma unroll
    for (int i = 0; i < FPL; ++i) {
        atomicAdd(dl + i, v * xl[i]);
        atomicAdd(dh + i, v * xh[i]);
    }
}

// ---------------------------------------------------------------------------
// Layer-1 attention + blend. One wave per node; emits bf16 h for layer-2 GEMM.
// XW rows: [xl(128) | xh(128) | xm(128)]
// ---------------------------------------------------------------------------
__global__ __launch_bounds__(256) void att_combine_l1(
    const float* __restrict__ accL, const float* __restrict__ accH,
    const float* __restrict__ XW,
    const float* __restrict__ aL, const float* __restrict__ aH,
    const float* __restrict__ aM, const float* __restrict__ aV,
    unsigned short* __restrict__ hb, int Nn)
{
    const int node = blockIdx.x * 8 + (threadIdx.x >> 5);
    if (node >= Nn) return;
    const int lane = threadIdx.x & 31;
    const int f    = lane * 4;

    const float4 zl = *reinterpret_cast<const float4*>(accL + (size_t)node * 128 + f);
    const float4 zh = *reinterpret_cast<const float4*>(accH + (size_t)node * 128 + f);
    const float4 xh = *reinterpret_cast<const float4*>(XW + (size_t)node * 384 + 128 + f);
    const float4 xm = *reinterpret_cast<const float4*>(XW + (size_t)node * 384 + 256 + f);

    float ol[4] = { fmaxf(zl.x, 0.f), fmaxf(zl.y, 0.f), fmaxf(zl.z, 0.f), fmaxf(zl.w, 0.f) };
    float oh[4] = { fmaxf(xh.x - zh.x, 0.f), fmaxf(xh.y - zh.y, 0.f),
                    fmaxf(xh.z - zh.z, 0.f), fmaxf(xh.w - zh.w, 0.f) };
    float om[4] = { fmaxf(xm.x, 0.f), fmaxf(xm.y, 0.f), fmaxf(xm.z, 0.f), fmaxf(xm.w, 0.f) };

    const float4 vl = *reinterpret_cast<const float4*>(aL + f);
    const float4 vh = *reinterpret_cast<const float4*>(aH + f);
    const float4 vm = *reinterpret_cast<const float4*>(aM + f);

    float dl = ol[0]*vl.x + ol[1]*vl.y + ol[2]*vl.z + ol[3]*vl.w;
    float dh = oh[0]*vh.x + oh[1]*vh.y + oh[2]*vh.z + oh[3]*vh.w;
    float dm = om[0]*vm.x + om[1]*vm.y + om[2]*vm.z + om[3]*vm.w;
    dl = waveSum(dl); dh = waveSum(dh); dm = waveSum(dm);

    const float s0 = 1.f / (1.f + __expf(-dl));
    const float s1 = 1.f / (1.f + __expf(-dh));
    const float s2 = 1.f / (1.f + __expf(-dm));
    const float t0 = (s0 * aV[0] + s1 * aV[3] + s2 * aV[6]) * (1.f / 3.f);
    const float t1 = (s0 * aV[1] + s1 * aV[4] + s2 * aV[7]) * (1.f / 3.f);
    const float t2 = (s0 * aV[2] + s1 * aV[5] + s2 * aV[8]) * (1.f / 3.f);
    const float mx = fmaxf(t0, fmaxf(t1, t2));
    const float e0 = __expf(t0 - mx), e1 = __expf(t1 - mx), e2 = __expf(t2 - mx);
    const float inv = 1.f / (e0 + e1 + e2);
    const float al = 3.f * e0 * inv, ah = 3.f * e1 * inv, am = 3.f * e2 * inv;

    unsigned short o[4];
#pragma unroll
    for (int i = 0; i < 4; ++i)
        o[i] = f2bf(al * ol[i] + ah * oh[i] + am * om[i]);
    const unsigned int p0 = (unsigned)o[0] | ((unsigned)o[1] << 16);
    const unsigned int p1 = (unsigned)o[2] | ((unsigned)o[3] << 16);
    *reinterpret_cast<uint2*>(hb + (size_t)node * 128 + f) = make_uint2(p0, p1);
}

// ---------------------------------------------------------------------------
// Layer-2 attention + blend + log_softmax(axis=1). One wave per node (C=64).
// XW2 rows: [xl(64) | xh(64) | xm(64)]
// ---------------------------------------------------------------------------
__global__ __launch_bounds__(256) void att_combine_l2_logsoftmax(
    const float* __restrict__ accL, const float* __restrict__ accH,
    const float* __restrict__ XW,
    const float* __restrict__ aL, const float* __restrict__ aH,
    const float* __restrict__ aM, const float* __restrict__ aV,
    float* __restrict__ out, int Nn)
{
    const int node = blockIdx.x * 8 + (threadIdx.x >> 5);
    if (node >= Nn) return;
    const int lane = threadIdx.x & 31;
    const int f    = lane * 2;

    const float2 zl = *reinterpret_cast<const float2*>(accL + (size_t)node * 64 + f);
    const float2 zh = *reinterpret_cast<const float2*>(accH + (size_t)node * 64 + f);
    const float2 xh = *reinterpret_cast<const float2*>(XW + (size_t)node * 192 + 64  + f);
    const float2 xm = *reinterpret_cast<const float2*>(XW + (size_t)node * 192 + 128 + f);

    float ol[2] = { fmaxf(zl.x, 0.f), fmaxf(zl.y, 0.f) };
    float oh[2] = { fmaxf(xh.x - zh.x, 0.f), fmaxf(xh.y - zh.y, 0.f) };
    float om[2] = { fmaxf(xm.x, 0.f), fmaxf(xm.y, 0.f) };

    const float2 vl = *reinterpret_cast<const float2*>(aL + f);
    const float2 vh = *reinterpret_cast<const float2*>(aH + f);
    const float2 vm = *reinterpret_cast<const float2*>(aM + f);

    float dl = waveSum(ol[0]*vl.x + ol[1]*vl.y);
    float dh = waveSum(oh[0]*vh.x + oh[1]*vh.y);
    float dm = waveSum(om[0]*vm.x + om[1]*vm.y);

    const float s0 = 1.f / (1.f + __expf(-dl));
    const float s1 = 1.f / (1.f + __expf(-dh));
    const float s2 = 1.f / (1.f + __expf(-dm));
    const float t0 = (s0 * aV[0] + s1 * aV[3] + s2 * aV[6]) * (1.f / 3.f);
    const float t1 = (s0 * aV[1] + s1 * aV[4] + s2 * aV[7]) * (1.f / 3.f);
    const float t2 = (s0 * aV[2] + s1 * aV[5] + s2 * aV[8]) * (1.f / 3.f);
    const float mx3 = fmaxf(t0, fmaxf(t1, t2));
    const float e0 = __expf(t0 - mx3), e1 = __expf(t1 - mx3), e2 = __expf(t2 - mx3);
    const float inv = 1.f / (e0 + e1 + e2);
    const float al = 3.f * e0 * inv, ah = 3.f * e1 * inv, am = 3.f * e2 * inv;

    const float c0 = al * ol[0] + ah * oh[0] + am * om[0];
    const float c1 = al * ol[1] + ah * oh[1] + am * om[1];

    const float mx = waveMax(fmaxf(c0, c1));
    const float se = waveSum(__expf(c0 - mx) + __expf(c1 - mx));
    const float ls = __logf(se);
    float2 r; r.x = c0 - mx - ls; r.y = c1 - mx - ls;
    *reinterpret_cast<float2*>(out + (size_t)node * 64 + f) = r;
}

// ---------------------------------------------------------------------------
// Host-side orchestration (all async on `stream`; scratch from d_ws only)
// ---------------------------------------------------------------------------
extern "C" void kernel_launch(void* const* d_in, const int* in_sizes, int n_in,
                              void* d_out, int out_size, void* d_ws, size_t ws_size,
                              hipStream_t stream) {
    (void)n_in; (void)out_size; (void)ws_size;

    const float* x    = (const float*)d_in[0];
    const int*   eRow = (const int*)d_in[1];
    const int*   eCol = (const int*)d_in[2];
    const float* eVal = (const float*)d_in[3];
    const float* Wl   = (const float*)d_in[4];
    const float* Wh   = (const float*)d_in[5];
    const float* Wm   = (const float*)d_in[6];
    const float* aL   = (const float*)d_in[7];
    const float* aH   = (const float*)d_in[8];
    const float* aM   = (const float*)d_in[9];
    const float* aV   = (const float*)d_in[10];
    const float* Wl2  = (const float*)d_in[11];
    const float* Wh2  = (const float*)d_in[12];
    const float* Wm2  = (const float*)d_in[13];
    const float* aL2  = (const float*)d_in[14];
    const float* aH2  = (const float*)d_in[15];
    const float* aM2  = (const float*)d_in[16];
    const float* aV2  = (const float*)d_in[17];
    float* out = (float*)d_out;

    const int FIN = 512, H = 128, Cc = 64;
    const int Nn = in_sizes[0] / FIN;
    const int E  = in_sizes[1];

    // workspace layout (256B-aligned slabs)
    char*  ws  = (char*)d_ws;
    size_t off = 0;
    auto take = [&](size_t bytes) {
        size_t o = off;
        off += (bytes + 255) & ~(size_t)255;
        return o;
    };
    unsigned short* Xb  = (unsigned short*)(ws + take((size_t)Nn * FIN * 2));
    unsigned short* hb  = (unsigned short*)(ws + take((size_t)Nn * H * 2));
    unsigned short* Wc1 = (unsigned short*)(ws + take((size_t)FIN * 3 * H * 2));
    unsigned short* Wc2 = (unsigned short*)(ws + take((size_t)H * 3 * Cc * 2));
    float* XW1   = (float*)(ws + take((size_t)Nn * 3 * H * 4));
    float* XW2   = (float*)(ws + take((size_t)Nn * 3 * Cc * 4));
    float* acc1  = (float*)(ws + take((size_t)2 * Nn * H * 4));
    float* acc2  = (float*)(ws + take((size_t)2 * Nn * Cc * 4));
    float* accL1 = acc1;  float* accH1 = acc1 + (size_t)Nn * H;
    float* accL2 = acc2;  float* accH2 = acc2 + (size_t)Nn * Cc;

    auto blocks = [](long n) { return (unsigned)((n + 255) / 256); };

    // ---- precision staging ----
    cvt_f32_bf16<<<blocks((long)Nn * FIN), 256, 0, stream>>>(x, Xb, (long)Nn * FIN);
    pack3_bf16<<<blocks((long)FIN * 3 * H), 256, 0, stream>>>(Wl, Wh, Wm, Wc1, FIN, H);
    pack3_bf16<<<blocks((long)H * 3 * Cc), 256, 0, stream>>>(Wl2, Wh2, Wm2, Wc2, H, Cc);
    zero_f32<<<blocks((long)2 * Nn * H), 256, 0, stream>>>(acc1, (long)2 * Nn * H);
    zero_f32<<<blocks((long)2 * Nn * Cc), 256, 0, stream>>>(acc2, (long)2 * Nn * Cc);

    // ---- layer 1: fused GEMM X @ [Wl|Wh|Wm]  (M=N, K=512, N=384) ----
    {
        dim3 grid((Nn + 127) / 128, (3 * H + 127) / 128);
        gemm_bf16_wmma<<<grid, 256, 0, stream>>>(Xb, Wc1, XW1, Nn, 3 * H, FIN);
    }
    spmm_dual<128><<<blocks((long)E * 32), 256, 0, stream>>>(
        eRow, eCol, eVal, XW1, 3 * H, H, accL1, accH1, E);
    att_combine_l1<<<(Nn + 7) / 8, 256, 0, stream>>>(
        accL1, accH1, XW1, aL, aH, aM, aV, hb, Nn);

    // ---- layer 2: fused GEMM h @ [Wl2|Wh2|Wm2]  (M=N, K=128, N=192) ----
    {
        dim3 grid((Nn + 127) / 128, (3 * Cc + 127) / 128);
        gemm_bf16_wmma<<<grid, 256, 0, stream>>>(hb, Wc2, XW2, Nn, 3 * Cc, H);
    }
    spmm_dual<64><<<blocks((long)E * 32), 256, 0, stream>>>(
        eRow, eCol, eVal, XW2, 3 * Cc, Cc, accL2, accH2, E);
    att_combine_l2_logsoftmax<<<(Nn + 7) / 8, 256, 0, stream>>>(
        accL2, accH2, XW2, aL2, aH2, aM2, aV2, out, Nn);
}
